// Downsample_60533269069907
// MI455X (gfx1250) — compile-verified, matrix operand order checked by reading, not run on hardware
//
#include <hip/hip_runtime.h>

// ---------------------------------------------------------------------------
// CDNA5 (gfx1250, wave32) implementation of the Downsample reference.
// Heavy GEMMs -> V_WMMA_F32_16X16X32_F16 (f32 accumulate).
// Edge-kernel W1 staging -> TENSOR_LOAD_TO_LDS (TDM) + s_wait_tensorcnt.
// ---------------------------------------------------------------------------

typedef __attribute__((ext_vector_type(16))) _Float16     v16h;
typedef __attribute__((ext_vector_type(8)))  float        v8f;
typedef __attribute__((ext_vector_type(4)))  unsigned int u32x4;
typedef __attribute__((ext_vector_type(8)))  int          i32x8;
typedef __attribute__((ext_vector_type(4)))  int          i32x4;

union Frag16 { v16h v; uint4 q[2]; };

#define NNODE   20000
#define HID     512
#define NUP     5000
#define NDOWN   15000
#define MKNN    5
#define KKNN    7
#define NEDGE   (NDOWN * MKNN)          // 75000
#define KEMB    515
#define KEMB_P  544                     // padded to 32
#define BNEPS   1e-5f

#if defined(__has_builtin)
#if __has_builtin(__builtin_amdgcn_tensor_load_to_lds) && __has_builtin(__builtin_amdgcn_s_wait_tensorcnt)
#define HAVE_TDM 1
#endif
#endif
#ifndef HAVE_TDM
#define HAVE_TDM 0
#endif

// K-index mapping for the 16-bit 16x32 A / 32x16 B WMMA fragments
// (lanes 0-15 carry K {0..7,16..23}; lanes 16-31 carry K {8..15,24..31})
__device__ __forceinline__ int kmap(int lane, int e) {
    int base = (lane >> 4) ? 8 : 0;
    return (e < 8) ? (base + e) : (16 + base + (e - 8));
}

__device__ __forceinline__ void atomicMaxFloat(float* addr, float val) {
    if (__float_as_int(val) >= 0)
        atomicMax((int*)addr, __float_as_int(val));
    else
        atomicMin((unsigned int*)addr, __float_as_uint(val));
}

#if HAVE_TDM
// TDM: 2D row-major tile (f16) global -> LDS.  D# packed per ISA 8.3/8.4:
//   g0: count=1 | lds_addr | global_addr[56:0] | type=2
//   g1: data_size=1(2B) | tensor_dim0/1 | tile_dim0/1 | tensor_dim0_stride
__device__ __forceinline__ void tdm_load_2d_f16(unsigned lds_off, const void* gptr,
                                                unsigned tile_cols, unsigned tile_rows,
                                                unsigned tensor_cols, unsigned tensor_rows,
                                                unsigned row_stride_elems) {
    unsigned long long ga = (unsigned long long)(size_t)gptr;
    u32x4 g0;
    g0[0] = 1u;                                           // count=1, normal mode
    g0[1] = lds_off;                                      // lds_addr (bytes)
    g0[2] = (unsigned)(ga & 0xFFFFFFFFu);                 // global_addr[31:0]
    g0[3] = (unsigned)((ga >> 32) & 0x01FFFFFFu) | (2u << 30);  // addr[56:32] | type=2
    i32x8 g1;
    g1[0] = (int)(1u << 16);                              // data_size=1 (2 bytes)
    g1[1] = (int)((tensor_cols & 0xFFFFu) << 16);         // ab_addr=0 | dim0[15:0]
    g1[2] = (int)(((tensor_cols >> 16) & 0xFFFFu) |
                  ((tensor_rows & 0xFFFFu) << 16));       // dim0[31:16] | dim1[15:0]
    g1[3] = (int)(((tensor_rows >> 16) & 0xFFFFu) |
                  ((tile_cols & 0xFFFFu) << 16));         // dim1[31:16] | tile_dim0
    g1[4] = (int)(tile_rows & 0xFFFFu);                   // tile_dim1 | tile_dim2=0
    g1[5] = (int)row_stride_elems;                        // dim0_stride[31:0]
    g1[6] = 0;                                            // stride hi | dim1_stride lo
    g1[7] = 0;
    i32x4 z4 = {0, 0, 0, 0};
#if defined(__clang_major__) && (__clang_major__ >= 23)
    i32x8 z8 = {0, 0, 0, 0, 0, 0, 0, 0};
    __builtin_amdgcn_tensor_load_to_lds(g0, g1, z4, z4, z8, 0);
#else
    __builtin_amdgcn_tensor_load_to_lds(g0, g1, z4, z4, 0);
#endif
}
#endif

// ---------------------------------------------------------------------------
// 1) top-25% selection: descending stable rank (ties -> larger index first,
//    matching argsort(ascending, stable)[::-1])
// ---------------------------------------------------------------------------
__global__ void k_rank(const float* __restrict__ scores, int n, int n_up,
                       int* __restrict__ mask) {
    int i = blockIdx.x * blockDim.x + threadIdx.x;
    if (i >= n) return;
    float si = scores[i];
    int cnt = 0;
    for (int j = 0; j < n; ++j) {
        float sj = scores[j];
        cnt += (sj > si) || (sj == si && j > i);
    }
    mask[i] = (cnt < n_up) ? 1 : 0;
}

// boolean-mask-order (ascending node id) compaction
__global__ void k_scatter(const int* __restrict__ mask, int n,
                          int* __restrict__ up_nodes, int* __restrict__ down_nodes) {
    int i = blockIdx.x * blockDim.x + threadIdx.x;
    if (i >= n) return;
    int c = 0;
    for (int j = 0; j < i; ++j) c += mask[j];
    if (mask[i]) up_nodes[c] = i;
    else         down_nodes[i - c] = i;
}

__global__ void k_pack_coords(const float* __restrict__ s_l, const int* __restrict__ nodes,
                              int n, float* __restrict__ xyz) {
    int i = blockIdx.x * blockDim.x + threadIdx.x;
    if (i >= n) return;
    int nd = nodes[i];
    xyz[i * 3 + 0] = s_l[(size_t)nd * 3 + 0];
    xyz[i * 3 + 1] = s_l[(size_t)nd * 3 + 1];
    xyz[i * 3 + 2] = s_l[(size_t)nd * 3 + 2];
}

// ---------------------------------------------------------------------------
// 2) KNN: down -> 5 nearest up; up -> 7 nearest up (self excluded)
// ---------------------------------------------------------------------------
__global__ void k_knn_down(const float* __restrict__ dxyz, const float* __restrict__ uxyz,
                           const int* __restrict__ down_nodes, const int* __restrict__ up_nodes,
                           int* __restrict__ ei, int* __restrict__ ej,
                           float* __restrict__ out_ei, float* __restrict__ out_ej) {
    int d = blockIdx.x * blockDim.x + threadIdx.x;
    if (d >= NDOWN) return;
    float x = dxyz[d * 3 + 0], y = dxyz[d * 3 + 1], z = dxyz[d * 3 + 2];
    float best[MKNN]; int bidx[MKNN];
#pragma unroll
    for (int j = 0; j < MKNN; ++j) { best[j] = 3.4e38f; bidx[j] = 0; }
    for (int u = 0; u < NUP; ++u) {
        float dx = x - uxyz[u * 3 + 0];
        float dy = y - uxyz[u * 3 + 1];
        float dz = z - uxyz[u * 3 + 2];
        float d2 = dx * dx + dy * dy + dz * dz;
        if (d2 < best[MKNN - 1]) {
            int pos = MKNN - 1;
            while (pos > 0 && d2 < best[pos - 1]) {
                best[pos] = best[pos - 1]; bidx[pos] = bidx[pos - 1]; --pos;
            }
            best[pos] = d2; bidx[pos] = u;
        }
    }
    int src = down_nodes[d];
#pragma unroll
    for (int j = 0; j < MKNN; ++j) {
        int e = d * MKNN + j;
        int dst = up_nodes[bidx[j]];
        ei[e] = src; ej[e] = dst;
        out_ei[e] = (float)src; out_ej[e] = (float)dst;
    }
}

__global__ void k_knn_up(const float* __restrict__ uxyz,
                         float* __restrict__ out_src, float* __restrict__ out_dst) {
    int i = blockIdx.x * blockDim.x + threadIdx.x;
    if (i >= NUP) return;
    float x = uxyz[i * 3 + 0], y = uxyz[i * 3 + 1], z = uxyz[i * 3 + 2];
    float best[KKNN]; int bidx[KKNN];
#pragma unroll
    for (int j = 0; j < KKNN; ++j) { best[j] = 3.4e38f; bidx[j] = 0; }
    for (int u = 0; u < NUP; ++u) {
        if (u == i) continue;
        float dx = x - uxyz[u * 3 + 0];
        float dy = y - uxyz[u * 3 + 1];
        float dz = z - uxyz[u * 3 + 2];
        float d2 = dx * dx + dy * dy + dz * dz;
        if (d2 < best[KKNN - 1]) {
            int pos = KKNN - 1;
            while (pos > 0 && d2 < best[pos - 1]) {
                best[pos] = best[pos - 1]; bidx[pos] = bidx[pos - 1]; --pos;
            }
            best[pos] = d2; bidx[pos] = u;
        }
    }
#pragma unroll
    for (int j = 0; j < KKNN; ++j) {
        out_src[i * KKNN + j] = (float)bidx[j];
        out_dst[i * KKNN + j] = (float)i;
    }
}

// ---------------------------------------------------------------------------
// 3) conversions
// ---------------------------------------------------------------------------
__global__ void k_cvt16(const float* __restrict__ W, _Float16* __restrict__ Wh,
                        int Korig, int Kpad, int n) {
    int idx = blockIdx.x * blockDim.x + threadIdx.x;
    if (idx >= Kpad * n) return;
    int k = idx / n, c = idx - k * n;
    Wh[idx] = (k < Korig) ? (_Float16)W[(size_t)k * n + c] : (_Float16)0.f;
}

__global__ void k_build_A_emb(const float* __restrict__ h, const float* __restrict__ s_l,
                              _Float16* __restrict__ A) {
    int idx = blockIdx.x * blockDim.x + threadIdx.x;
    if (idx >= NNODE * KEMB_P) return;
    int r = idx / KEMB_P, k = idx - r * KEMB_P;
    float v = 0.f;
    if (k < HID)            v = h[(size_t)r * HID + k];
    else if (k < KEMB)      v = s_l[(size_t)r * 3 + (k - HID)];
    A[idx] = (_Float16)v;
}

// ---------------------------------------------------------------------------
// 4) generic WMMA GEMM: out = act(A[MxK]f16 @ B[KxN]f16 + bias (+resid))
//    block: 256 thr = 8 waves, tile 128 rows x 16 cols; B tile staged in LDS
//    in WMMA fragment order, A fragment loaded 16B-vectorized from global.
// ---------------------------------------------------------------------------
__global__ __launch_bounds__(256)
void k_gemm(const _Float16* __restrict__ A, int K,
            const _Float16* __restrict__ B, int N,
            const float* __restrict__ bias,
            const float* __restrict__ resid,
            float* __restrict__ out32,
            _Float16* __restrict__ out16,
            int M, int relu) {
    __shared__ __align__(16) _Float16 sB[512];   // fragment-order 32x16 tile
    const int lane = threadIdx.x & 31;
    const int wave = threadIdx.x >> 5;
    const int hi   = lane >> 4;
    const int n0   = blockIdx.x * 16;
    const int m0   = blockIdx.y * 128 + wave * 16;
    const bool wv  = (m0 < M);                    // wave-uniform

    v8f acc = {};
    const int arow = m0 + (lane & 15);

    for (int k0 = 0; k0 < K; k0 += 32) {
        // cooperative B stage: sB[(lane,e)] = B[k0+kmap(lane,e)][n0 + lane%16]
#pragma unroll
        for (int it = 0; it < 2; ++it) {
            int idx = threadIdx.x + it * 256;
            int l = idx >> 4, e = idx & 15;
            sB[idx] = B[(size_t)(k0 + kmap(l, e)) * N + n0 + (l & 15)];
        }
        __syncthreads();
        if (wv) {
            Frag16 a, b;
            const _Float16* ap = A + (size_t)arow * K + k0 + hi * 8;
            a.q[0] = *reinterpret_cast<const uint4*>(ap);
            a.q[1] = *reinterpret_cast<const uint4*>(ap + 16);
            __builtin_prefetch(ap + 32, 0, 1);    // global_prefetch_b8
            b.q[0] = *reinterpret_cast<const uint4*>(&sB[lane * 16]);
            b.q[1] = *reinterpret_cast<const uint4*>(&sB[lane * 16 + 8]);
            acc = __builtin_amdgcn_wmma_f32_16x16x32_f16(
                false, a.v, false, b.v, (short)0, acc, false, false);
        }
        __syncthreads();
    }
    if (wv) {
        int col = n0 + (lane & 15);
        float bc = bias ? bias[col] : 0.f;
#pragma unroll
        for (int r = 0; r < 8; ++r) {
            int rr = m0 + r + hi * 8;             // D layout: M = r + 8*hi
            float v = acc[r] + bc;
            if (resid) v += resid[(size_t)rr * N + col];
            if (relu)  v = fmaxf(v, 0.f);
            if (out32) out32[(size_t)rr * N + col] = v;
            if (out16) out16[(size_t)rr * N + col] = (_Float16)v;
        }
    }
}

// ---------------------------------------------------------------------------
// 5) fused EdgeDistancesPassing + MeanMax scatter.
//    One block per 16-edge tile. D = (feats[ei]-feats[ej]) @ W1 kept in regs
//    (8 waves x 4 col-tiles x v8f), att = relu(relu(D+b1)@W2 + b2),
//    msg = exp(-att)*feats[ej] scattered with global float atomics.
//    W1 K-slice (32x512 f16) staged into LDS by the Tensor Data Mover.
// ---------------------------------------------------------------------------
__global__ __launch_bounds__(256)
void k_edge_att(const float* __restrict__ feats32,
                const _Float16* __restrict__ W1h,     // 512x512
                const float* __restrict__ b1,
                const float* __restrict__ W2,         // 512
                const float* __restrict__ b2,         // 1
                const int* __restrict__ ei, const int* __restrict__ ej,
                float* __restrict__ sum_agg, float* __restrict__ max_agg,
                float* __restrict__ deg) {
    __shared__ __align__(16) _Float16 sA[512];        // dif fragment (16x32)
    __shared__ __align__(16) _Float16 sW[32 * 512];   // W1 K-slice, row-major
    __shared__ float sAtt[16];
    __shared__ float sWgt[16];

    const int tid  = threadIdx.x;
    const int lane = tid & 31;
    const int wave = tid >> 5;
    const int hi   = lane >> 4;
    const int et   = blockIdx.x;

    if (tid < 16) sAtt[tid] = 0.f;

#if HAVE_TDM
    const unsigned sW_lds = (unsigned)(size_t)(void*)&sW[0];
#endif

    v8f acc0 = {}, acc1 = {}, acc2 = {}, acc3 = {};

    for (int k0 = 0; k0 < HID; k0 += 32) {
        // stage dif fragment (clamped edge index for the ragged last tile)
#pragma unroll
        for (int it = 0; it < 2; ++it) {
            int idx = tid + it * 256;
            int l = idx >> 4, e = idx & 15;
            int kk = kmap(l, e);
            int edge = et * 16 + (l & 15);
            if (edge >= NEDGE) edge = NEDGE - 1;
            float dv = feats32[(size_t)ei[edge] * HID + k0 + kk]
                     - feats32[(size_t)ej[edge] * HID + k0 + kk];
            sA[idx] = (_Float16)dv;
        }
#if HAVE_TDM
        // TDM: async DMA of W1 rows [k0, k0+32) x 512 cols into LDS
        if (wave == 0) {                              // wave-uniform; EXEC ignored by TDM
            tdm_load_2d_f16(sW_lds, W1h + (size_t)k0 * HID,
                            /*tile_cols=*/HID, /*tile_rows=*/32,
                            /*tensor_cols=*/HID, /*tensor_rows=*/32,
                            /*row_stride=*/HID);
            __builtin_amdgcn_s_wait_tensorcnt(0);
        }
#else
        {
            const uint4* src = reinterpret_cast<const uint4*>(W1h + (size_t)k0 * HID);
            uint4* dst = reinterpret_cast<uint4*>(sW);
#pragma unroll
            for (int it = 0; it < 8; ++it) dst[tid + it * 256] = src[tid + it * 256];
        }
#endif
        __syncthreads();
        Frag16 a;
        a.q[0] = *reinterpret_cast<const uint4*>(&sA[lane * 16]);
        a.q[1] = *reinterpret_cast<const uint4*>(&sA[lane * 16 + 8]);
#pragma unroll
        for (int t = 0; t < 4; ++t) {
            Frag16 b;
            int col = wave * 64 + t * 16 + (lane & 15);
#pragma unroll
            for (int e = 0; e < 16; ++e)
                b.v[e] = sW[kmap(lane, e) * HID + col];
            v8f* ap = (t == 0) ? &acc0 : (t == 1) ? &acc1 : (t == 2) ? &acc2 : &acc3;
            *ap = __builtin_amdgcn_wmma_f32_16x16x32_f16(
                false, a.v, false, b.v, (short)0, *ap, false, false);
        }
        __syncthreads();
    }

    // relu(D + b1) . W2, reduced per edge-row via LDS float atomics
    float part[8];
#pragma unroll
    for (int r = 0; r < 8; ++r) part[r] = 0.f;
#pragma unroll
    for (int t = 0; t < 4; ++t) {
        int col = wave * 64 + t * 16 + (lane & 15);
        float bb = b1[col], w2 = W2[col];
        const v8f& ac = (t == 0) ? acc0 : (t == 1) ? acc1 : (t == 2) ? acc2 : acc3;
#pragma unroll
        for (int r = 0; r < 8; ++r)
            part[r] += fmaxf(ac[r] + bb, 0.f) * w2;
    }
#pragma unroll
    for (int r = 0; r < 8; ++r)
        atomicAdd(&sAtt[r + hi * 8], part[r]);      // ds_add_f32
    __syncthreads();
    if (tid < 16) {
        float att = fmaxf(sAtt[tid] + b2[0], 0.f);
        sWgt[tid] = __expf(-att);
        int edge = et * 16 + tid;
        if (edge < NEDGE) atomicAdd(&deg[ej[edge]], 1.0f);
    }
    __syncthreads();

    // scatter msg = w * feats[ej] into sum/max aggregators
    for (int idx = tid; idx < 16 * HID; idx += 256) {
        int m = idx >> 9, f = idx & (HID - 1);
        int edge = et * 16 + m;
        if (edge < NEDGE) {
            int dst = ej[edge];
            float msg = sWgt[m] * feats32[(size_t)dst * HID + f];
            atomicAdd(&sum_agg[(size_t)dst * HID + f], msg);
            atomicMaxFloat(&max_agg[(size_t)dst * HID + f], msg);
        }
    }
}

// ---------------------------------------------------------------------------
// 6) aggregation buffers init / finalize
// ---------------------------------------------------------------------------
__global__ void k_init_agg(float* __restrict__ sum_agg, float* __restrict__ max_agg,
                           float* __restrict__ deg) {
    int i = blockIdx.x * blockDim.x + threadIdx.x;
    if (i < NNODE * HID) {
        sum_agg[i] = 0.f;
        max_agg[i] = __int_as_float(0xFF800000);  // -inf
    }
    if (i < NNODE) deg[i] = 0.f;
}

__global__ void k_finalize_agg(const float* __restrict__ sum_agg,
                               const float* __restrict__ max_agg,
                               const float* __restrict__ deg,
                               _Float16* __restrict__ agg_h) {
    int idx = blockIdx.x * blockDim.x + threadIdx.x;
    if (idx >= NNODE * HID) return;
    int n = idx >> 9, c = idx & (HID - 1);
    float d = deg[n];
    float mn = 0.f, mx = 0.f;
    if (d > 0.f) { mn = sum_agg[idx] / d; mx = max_agg[idx]; }
    agg_h[(size_t)n * (2 * HID) + c]       = (_Float16)mn;
    agg_h[(size_t)n * (2 * HID) + HID + c] = (_Float16)mx;
}

// ---------------------------------------------------------------------------
// 7) BatchNorm (training mode, biased variance)
// ---------------------------------------------------------------------------
__global__ __launch_bounds__(256)
void k_bn_stats(const float* __restrict__ X, int M, int N,
                float* __restrict__ mu, float* __restrict__ rs) {
    __shared__ float s1[256], s2[256];
    int c = blockIdx.x;
    float a = 0.f, b = 0.f;
    for (int r = threadIdx.x; r < M; r += 256) {
        float x = X[(size_t)r * N + c];
        a += x; b += x * x;
    }
    s1[threadIdx.x] = a; s2[threadIdx.x] = b;
    __syncthreads();
    for (int off = 128; off > 0; off >>= 1) {
        if (threadIdx.x < off) {
            s1[threadIdx.x] += s1[threadIdx.x + off];
            s2[threadIdx.x] += s2[threadIdx.x + off];
        }
        __syncthreads();
    }
    if (threadIdx.x == 0) {
        float m = s1[0] / (float)M;
        float v = s2[0] / (float)M - m * m;
        mu[c] = m;
        rs[c] = rsqrtf(fmaxf(v, 0.f) + BNEPS);
    }
}

__global__ void k_bn_apply(const float* __restrict__ X,
                           const float* __restrict__ mu, const float* __restrict__ rs,
                           const float* __restrict__ g, const float* __restrict__ be,
                           float* __restrict__ out32, _Float16* __restrict__ out16,
                           int M, int N) {
    int idx = blockIdx.x * blockDim.x + threadIdx.x;
    if (idx >= M * N) return;
    int c = idx % N;
    float v = (X[idx] - mu[c]) * rs[c] * g[c] + be[c];
    if (out32) out32[idx] = v;
    if (out16) out16[idx] = (_Float16)v;
}

__global__ void k_write_mask(const int* __restrict__ mask, float* __restrict__ om) {
    int i = blockIdx.x * blockDim.x + threadIdx.x;
    if (i < NNODE) om[i] = mask[i] ? 1.f : 0.f;
}

// ---------------------------------------------------------------------------
// launch
// ---------------------------------------------------------------------------
extern "C" void kernel_launch(void* const* d_in, const int* in_sizes, int n_in,
                              void* d_out, int out_size, void* d_ws, size_t ws_size,
                              hipStream_t stream) {
    (void)in_sizes; (void)n_in; (void)out_size; (void)ws_size;

    const float* h      = (const float*)d_in[0];
    const float* s_l    = (const float*)d_in[1];
    const float* scores = (const float*)d_in[2];
    const float* W_emb  = (const float*)d_in[3];
    const float* b_emb  = (const float*)d_in[4];
    const float* W1     = (const float*)d_in[5];
    const float* b1     = (const float*)d_in[6];
    const float* W2     = (const float*)d_in[7];
    const float* b2     = (const float*)d_in[8];
    const float* W_O    = (const float*)d_in[9];
    const float* b_O    = (const float*)d_in[10];
    const float* g1     = (const float*)d_in[11];
    const float* be1    = (const float*)d_in[12];
    const float* Wf1    = (const float*)d_in[13];
    const float* bf1    = (const float*)d_in[14];
    const float* Wf2    = (const float*)d_in[15];
    const float* bf2    = (const float*)d_in[16];
    const float* g2     = (const float*)d_in[17];
    const float* be2    = (const float*)d_in[18];

    float* out = (float*)d_out;
    float* o_out  = out;
    float* o_mask = out + (size_t)NNODE * HID;                 // 10,240,000
    float* o_ei   = o_mask + NNODE;
    float* o_ej   = o_ei + NEDGE;
    float* o_ks   = o_ej + NEDGE;
    float* o_kd   = o_ks + (size_t)NUP * KKNN;

    // ---- workspace carve ----
    char* p = (char*)d_ws;
    auto alloc = [&](size_t bytes) {
        char* r = p;
        p += (bytes + 255) & ~(size_t)255;
        return r;
    };
    float*    feats32 = (float*)   alloc((size_t)NNODE * HID * 4);
    _Float16* feats16 = (_Float16*)alloc((size_t)NNODE * HID * 2);
    _Float16* Aemb    = (_Float16*)alloc((size_t)NNODE * KEMB_P * 2);
    _Float16* Wembh   = (_Float16*)alloc((size_t)KEMB_P * HID * 2);
    _Float16* W1h     = (_Float16*)alloc((size_t)HID * HID * 2);
    _Float16* WOh     = (_Float16*)alloc((size_t)(2 * HID) * HID * 2);
    _Float16* Wf1h    = (_Float16*)alloc((size_t)HID * (2 * HID) * 2);
    _Float16* Wf2h    = (_Float16*)alloc((size_t)(2 * HID) * HID * 2);
    float*    sum_agg = (float*)   alloc((size_t)NNODE * HID * 4);  // reused: pre_bn1
    float*    max_agg = (float*)   alloc((size_t)NNODE * HID * 4);  // reused: pre_bn2
    _Float16* agg_h   = (_Float16*)alloc((size_t)NNODE * 2 * HID * 2); // reused: t16
    float*    hh32    = (float*)   alloc((size_t)NNODE * HID * 4);
    _Float16* hh16    = (_Float16*)alloc((size_t)NNODE * HID * 2);
    float*    deg     = (float*)   alloc((size_t)NNODE * 4);
    int*      mask_i  = (int*)     alloc((size_t)NNODE * 4);
    int*      up_n    = (int*)     alloc((size_t)NUP * 4);
    int*      dn_n    = (int*)     alloc((size_t)NDOWN * 4);
    int*      ei_i    = (int*)     alloc((size_t)NEDGE * 4);
    int*      ej_i    = (int*)     alloc((size_t)NEDGE * 4);
    float*    up_xyz  = (float*)   alloc((size_t)NUP * 3 * 4);
    float*    dn_xyz  = (float*)   alloc((size_t)NDOWN * 3 * 4);
    float*    mu      = (float*)   alloc(1024 * 4);
    float*    rs      = (float*)   alloc(1024 * 4);
    float*    pre_bn1 = sum_agg;
    float*    pre_bn2 = max_agg;
    _Float16* t16     = agg_h;

    // ---- selection + KNN graph ----
    k_rank      <<<(NNODE + 255) / 256, 256, 0, stream>>>(scores, NNODE, NUP, mask_i);
    k_scatter   <<<(NNODE + 255) / 256, 256, 0, stream>>>(mask_i, NNODE, up_n, dn_n);
    k_pack_coords<<<(NUP + 255) / 256, 256, 0, stream>>>(s_l, up_n, NUP, up_xyz);
    k_pack_coords<<<(NDOWN + 255) / 256, 256, 0, stream>>>(s_l, dn_n, NDOWN, dn_xyz);
    k_knn_down  <<<(NDOWN + 63) / 64, 64, 0, stream>>>(dn_xyz, up_xyz, dn_n, up_n,
                                                       ei_i, ej_i, o_ei, o_ej);
    k_knn_up    <<<(NUP + 63) / 64, 64, 0, stream>>>(up_xyz, o_ks, o_kd);
    k_write_mask<<<(NNODE + 255) / 256, 256, 0, stream>>>(mask_i, o_mask);

    // ---- f16 operand prep ----
    k_build_A_emb<<<(NNODE * KEMB_P + 255) / 256, 256, 0, stream>>>(h, s_l, Aemb);
    k_cvt16<<<(KEMB_P * HID + 255) / 256, 256, 0, stream>>>(W_emb, Wembh, KEMB, KEMB_P, HID);
    k_cvt16<<<(HID * HID + 255) / 256, 256, 0, stream>>>(W1, W1h, HID, HID, HID);
    k_cvt16<<<(2 * HID * HID + 255) / 256, 256, 0, stream>>>(W_O, WOh, 2 * HID, 2 * HID, HID);
    k_cvt16<<<(HID * 2 * HID + 255) / 256, 256, 0, stream>>>(Wf1, Wf1h, HID, HID, 2 * HID);
    k_cvt16<<<(2 * HID * HID + 255) / 256, 256, 0, stream>>>(Wf2, Wf2h, 2 * HID, 2 * HID, HID);

    const dim3 blk(256);
    const int  mgrid = (NNODE + 127) / 128;  // 157

    // ---- embedding GEMM: feats = [h|s_l] @ W_emb + b ----
    k_gemm<<<dim3(HID / 16, mgrid), blk, 0, stream>>>(
        Aemb, KEMB_P, Wembh, HID, b_emb, nullptr, feats32, feats16, NNODE, 0);

    // ---- fused per-edge MLP attention + mean/max scatter ----
    k_init_agg<<<(NNODE * HID + 255) / 256, 256, 0, stream>>>(sum_agg, max_agg, deg);
    k_edge_att<<<(NEDGE + 15) / 16, blk, 0, stream>>>(
        feats32, W1h, b1, W2, b2, ei_i, ej_i, sum_agg, max_agg, deg);
    k_finalize_agg<<<(NNODE * HID + 255) / 256, 256, 0, stream>>>(
        sum_agg, max_agg, deg, agg_h);

    // ---- upd = [mean|max] @ W_O + b_O + feats (residual) ----
    k_gemm<<<dim3(HID / 16, mgrid), blk, 0, stream>>>(
        agg_h, 2 * HID, WOh, HID, b_O, feats32, pre_bn1, nullptr, NNODE, 0);

    // ---- BN1 -> hh ----
    k_bn_stats<<<HID, blk, 0, stream>>>(pre_bn1, NNODE, HID, mu, rs);
    k_bn_apply<<<(NNODE * HID + 255) / 256, 256, 0, stream>>>(
        pre_bn1, mu, rs, g1, be1, hh32, hh16, NNODE, HID);

    // ---- FFN ----
    k_gemm<<<dim3(2 * HID / 16, mgrid), blk, 0, stream>>>(
        hh16, HID, Wf1h, 2 * HID, bf1, nullptr, nullptr, t16, NNODE, 1);
    k_gemm<<<dim3(HID / 16, mgrid), blk, 0, stream>>>(
        t16, 2 * HID, Wf2h, HID, bf2, hh32, pre_bn2, nullptr, NNODE, 0);

    // ---- BN2 -> out ----
    k_bn_stats<<<HID, blk, 0, stream>>>(pre_bn2, NNODE, HID, mu, rs);
    k_bn_apply<<<(NNODE * HID + 255) / 256, 256, 0, stream>>>(
        pre_bn2, mu, rs, g2, be2, o_out, nullptr, NNODE, HID);
}
